// StackedRNNLayer_50345606644268
// MI455X (gfx1250) — matrix operand order repaired and checked by reading, not compiled, not run on hardware
//
#include <hip/hip_runtime.h>
#include <stdint.h>

// ---------------- problem constants ----------------
#define T_STEPS 512
#define BATCH   32
#define DIM     1024          // D == H == O
#define G3      3072          // 3*H
#define M_TOT   (T_STEPS*BATCH)   // 16384
#define NWG     64            // persistent workgroups for the scan
#define SLICE   16            // hidden/output columns owned per WG

typedef __attribute__((ext_vector_type(16))) __bf16      v16bf;
typedef __attribute__((ext_vector_type(8)))  float       v8f;
typedef __attribute__((ext_vector_type(4)))  unsigned int u32x4;
typedef __attribute__((ext_vector_type(4)))  float       f32x4;

union Frag { u32x4 q[2]; v16bf v; };   // 32B: one 16x16x32 bf16 A or B fragment per lane

__device__ __forceinline__ unsigned short bf16r(float f) {
  unsigned int u = __float_as_uint(f);
  return (unsigned short)((u + 0x7FFFu + ((u >> 16) & 1u)) >> 16);
}
__device__ __forceinline__ unsigned int packbf(float lo, float hi) {
  return (unsigned int)bf16r(lo) | ((unsigned int)bf16r(hi) << 16);
}

// ---------------------------------------------------------------------------
// Kernel 0: transpose + convert recurrent weights to bf16, column-major [col][K]
// wT [3072][1024]  <- W_hm  [1024][3072]
// wpT[1024][1024]  <- W_proj[1024][1024]
// ---------------------------------------------------------------------------
__global__ void convert_weights(const float* __restrict__ Whm,
                                const float* __restrict__ Wproj,
                                unsigned short* __restrict__ wT,
                                unsigned short* __restrict__ wpT) {
  const int n1 = G3 * DIM;
  const int n2 = DIM * DIM;
  for (int idx = blockIdx.x * blockDim.x + threadIdx.x; idx < n1 + n2;
       idx += gridDim.x * blockDim.x) {
    if (idx < n1) {
      int c = idx / DIM, k = idx - c * DIM;
      wT[idx] = bf16r(Whm[(size_t)k * G3 + c]);
    } else {
      int j = idx - n1;
      int c = j / DIM, k = j - c * DIM;
      wpT[j] = bf16r(Wproj[(size_t)k * DIM + c]);
    }
  }
}

// ---------------------------------------------------------------------------
// Kernel 1: gates_x = A @ W_ih + b       A:[16384,1024] (f32 or bf16)
//                                        W:[1024,3072] f32,  out f32 [16384,3072]
// 128x128 block tile, BK=32, 8 waves (4Mx2N), double-buffered LDS, bf16 WMMA.
// ---------------------------------------------------------------------------
#define BK  32
#define STR 40   // padded LDS K-stride (elems); keeps 16B alignment for b128 frag reads

__global__ __launch_bounds__(256)
void gates_gemm(const float* __restrict__ Af32,
                const unsigned short* __restrict__ Abf, int a_is_bf,
                const float* __restrict__ W, const float* __restrict__ bias,
                float* __restrict__ out) {
  __shared__ unsigned short Alds[2][128 * STR];
  __shared__ unsigned short Blds[2][128 * STR];

  const int tid  = threadIdx.x;
  const int lane = tid & 31, wave = tid >> 5;
  const int hi = lane >> 4, ln = lane & 15;
  const int m0 = blockIdx.y * 128, n0 = blockIdx.x * 128;
  const int wm = wave >> 1, wn = wave & 1;       // wave tile: 32 rows x 64 cols

  // staging assignment
  const int ar = tid >> 1;           // A row 0..127
  const int as = (tid & 1) * 16;     // A k-segment {0,16}
  const int bc = tid & 127;          // B col 0..127
  const int bs = (tid >> 7) * 16;    // B k-segment {0,16}

  unsigned int regA[8], regB[8];
  v8f acc[2][4];
#pragma unroll
  for (int i = 0; i < 2; i++)
#pragma unroll
    for (int j = 0; j < 4; j++) acc[i][j] = (v8f){};

  auto loadA = [&](int ks) {
    size_t base = (size_t)(m0 + ar) * DIM + ks * BK + as;
    if (a_is_bf) {
      u32x4 x = *(const u32x4*)(Abf + base);
      u32x4 y = *(const u32x4*)(Abf + base + 8);
      regA[0]=x.x; regA[1]=x.y; regA[2]=x.z; regA[3]=x.w;
      regA[4]=y.x; regA[5]=y.y; regA[6]=y.z; regA[7]=y.w;
    } else {
      const f32x4* p = (const f32x4*)(Af32 + base);
      f32x4 f0 = p[0], f1 = p[1], f2 = p[2], f3 = p[3];
      regA[0]=packbf(f0.x,f0.y); regA[1]=packbf(f0.z,f0.w);
      regA[2]=packbf(f1.x,f1.y); regA[3]=packbf(f1.z,f1.w);
      regA[4]=packbf(f2.x,f2.y); regA[5]=packbf(f2.z,f2.w);
      regA[6]=packbf(f3.x,f3.y); regA[7]=packbf(f3.z,f3.w);
    }
  };
  auto loadB = [&](int ks) {
    const float* p = W + (size_t)(ks * BK + bs) * G3 + n0 + bc;
    float f[16];
#pragma unroll
    for (int i = 0; i < 16; i++) f[i] = p[(size_t)i * G3];
#pragma unroll
    for (int i = 0; i < 8; i++) regB[i] = packbf(f[2*i], f[2*i+1]);
  };
  auto storeLDS = [&](int buf) {
    unsigned int* ap = (unsigned int*)&Alds[buf][ar * STR + as];
#pragma unroll
    for (int i = 0; i < 8; i++) ap[i] = regA[i];
    unsigned int* bp = (unsigned int*)&Blds[buf][bc * STR + bs];
#pragma unroll
    for (int i = 0; i < 8; i++) bp[i] = regB[i];
  };
  auto compute = [&](int buf) {
    Frag a[2], b[4];
#pragma unroll
    for (int mt = 0; mt < 2; mt++) {
      int row = wm * 32 + mt * 16 + ln;
      a[mt].q[0] = *(const u32x4*)&Alds[buf][row * STR + hi * 16];        // K 0..7 (+kbase)
      a[mt].q[1] = *(const u32x4*)&Alds[buf][row * STR + hi * 16 + 8];    // K 8..15
    }
#pragma unroll
    for (int nt = 0; nt < 4; nt++) {
      int col = wn * 64 + nt * 16 + ln;
      const u32x4* p = (const u32x4*)&Blds[buf][col * STR + hi * 16];
      b[nt].q[0] = p[0]; b[nt].q[1] = p[1];
    }
#pragma unroll
    for (int mt = 0; mt < 2; mt++)
#pragma unroll
      for (int nt = 0; nt < 4; nt++)
        acc[mt][nt] = __builtin_amdgcn_wmma_f32_16x16x32_bf16(
            false, a[mt].v, false, b[nt].v, (short)0, acc[mt][nt], false, false);
  };

  loadA(0); loadB(0); storeLDS(0);
  __syncthreads();
#pragma unroll 1
  for (int ks = 0; ks < 32; ks++) {
    if (ks < 31) { loadA(ks + 1); loadB(ks + 1); }
    compute(ks & 1);
    if (ks < 31) storeLDS((ks + 1) & 1);
    __syncthreads();
  }

#pragma unroll
  for (int mt = 0; mt < 2; mt++)
#pragma unroll
    for (int nt = 0; nt < 4; nt++)
#pragma unroll
      for (int j = 0; j < 8; j++) {
        int row = m0 + wm * 32 + mt * 16 + hi * 8 + j;
        int col = n0 + wn * 64 + nt * 16 + ln;
        out[(size_t)row * G3 + col] = acc[mt][nt][j] + bias[col];
      }
}

// ---------------------------------------------------------------------------
// Grid barrier: monotonic counter in L2, release/acquire at agent scope.
// ---------------------------------------------------------------------------
__device__ __forceinline__ void grid_barrier(unsigned int* bar, unsigned int* nb) {
  __threadfence();
  __syncthreads();
  *nb += 1;
  if (threadIdx.x == 0) {
    unsigned int target = (*nb) * NWG;
    __hip_atomic_fetch_add(bar, 1u, __ATOMIC_ACQ_REL, __HIP_MEMORY_SCOPE_AGENT);
    while (__hip_atomic_load(bar, __ATOMIC_ACQUIRE, __HIP_MEMORY_SCOPE_AGENT) < target)
      __builtin_amdgcn_s_sleep(2);
  }
  __syncthreads();
  __threadfence();
}

// ---------------------------------------------------------------------------
// Kernel 2: persistent GRU scan. 64 WGs, each owns 16 hidden/output columns.
// Per step: A) r,u = sigmoid(gx + m@W)  -> rm      (grid barrier)
//           B) n = tanh(gx + rm@Wn), c update      (grid barrier)
//           C) m = pad-mix(c @ W_proj), emit ys    (grid barrier)
// ---------------------------------------------------------------------------
__global__ __launch_bounds__(256)
void gru_scan(const float* __restrict__ gbuf,            // [T*B, 3072]
              const float* __restrict__ padding,         // [T*B]
              const unsigned short* __restrict__ wT,     // [3072][1024] bf16
              const unsigned short* __restrict__ wpT,    // [1024][1024] bf16
              unsigned short* __restrict__ m_bf,         // [32][1024]
              unsigned short* __restrict__ rm_bf,        // [32][1024]
              unsigned short* __restrict__ c_bf,         // [32][1024]
              unsigned int* __restrict__ bar,
              float* __restrict__ outf,                  // final layer output
              unsigned short* __restrict__ outbf,        // intermediate ys (bf16)
              int out_is_bf) {
  __shared__ float partial[8][256];
  __shared__ float u_s[BATCH * SLICE];
  __shared__ float c_s[BATCH * SLICE];
  __shared__ float m_s[BATCH * SLICE];

  const int tid = threadIdx.x, lane = tid & 31, wave = tid >> 5;
  const int hi = lane >> 4, ln = lane & 15;
  const int g = blockIdx.x;
  const int hs0 = g * SLICE;

  for (int i = tid; i < BATCH * SLICE; i += 256) { u_s[i] = 0.f; c_s[i] = 0.f; m_s[i] = 0.f; }
  __syncthreads();
  unsigned int nb = 0;

  for (int t = 0; t < T_STEPS; t++) {
    const float* gx = gbuf + (size_t)t * BATCH * G3;

    // ---- Phase A: r,u gates (2 gates x 2 Mtiles x 2 K-halves over 8 waves) ----
    {
      const int gi = wave & 1, mt = (wave >> 1) & 1, kh = wave >> 2;
      const int arow = mt * 16 + ln;
      const int col = gi * DIM + hs0 + ln;
      v8f acc = (v8f){};
#pragma unroll
      for (int kk = 0; kk < 16; kk++) {
        int k0 = kh * 512 + kk * 32;
        Frag a, b;
        const unsigned short* ap = m_bf + (size_t)arow * DIM + k0 + hi * 16;
        a.q[0] = *(const u32x4*)ap;  a.q[1] = *(const u32x4*)(ap + 8);
        const unsigned short* bp = wT + (size_t)col * DIM + k0 + hi * 16;
        b.q[0] = ((const u32x4*)bp)[0];  b.q[1] = ((const u32x4*)bp)[1];
        acc = __builtin_amdgcn_wmma_f32_16x16x32_bf16(false, a.v, false, b.v,
                                                      (short)0, acc, false, false);
      }
#pragma unroll
      for (int j = 0; j < 8; j++) partial[wave][(j + hi * 8) * 16 + ln] = acc[j];
    }
    __syncthreads();
    for (int e = tid; e < 1024; e += 256) {   // r: e<512, u: e>=512
      int gi = e >> 9, idx = e & 511;
      int rw = idx >> 4, cl = idx & 15;
      int mt = rw >> 4, rit = (rw & 15) * 16 + cl;
      float s = partial[gi + 2 * mt][rit] + partial[gi + 2 * mt + 4][rit];
      float val = gx[(size_t)rw * G3 + gi * DIM + hs0 + cl] + s;
      float sg = 1.0f / (1.0f + __expf(-val));
      if (gi == 0) rm_bf[rw * DIM + hs0 + cl] = bf16r(sg * m_s[idx]);
      else         u_s[idx] = sg;
    }
    grid_barrier(bar, &nb);

    // ---- Phase B: n gate + c update (2 Mtiles x 4 K-quarters) ----
    {
      const int mt = wave & 1, kq = wave >> 1;
      const int arow = mt * 16 + ln;
      const int col = 2 * DIM + hs0 + ln;
      v8f acc = (v8f){};
#pragma unroll
      for (int kk = 0; kk < 8; kk++) {
        int k0 = kq * 256 + kk * 32;
        Frag a, b;
        const unsigned short* ap = rm_bf + (size_t)arow * DIM + k0 + hi * 16;
        a.q[0] = *(const u32x4*)ap;  a.q[1] = *(const u32x4*)(ap + 8);
        const unsigned short* bp = wT + (size_t)col * DIM + k0 + hi * 16;
        b.q[0] = ((const u32x4*)bp)[0];  b.q[1] = ((const u32x4*)bp)[1];
        acc = __builtin_amdgcn_wmma_f32_16x16x32_bf16(false, a.v, false, b.v,
                                                      (short)0, acc, false, false);
      }
#pragma unroll
      for (int j = 0; j < 8; j++) partial[wave][(j + hi * 8) * 16 + ln] = acc[j];
    }
    __syncthreads();
    for (int e = tid; e < 512; e += 256) {
      int rw = e >> 4, cl = e & 15;
      int mt = rw >> 4, rit = (rw & 15) * 16 + cl;
      float gn = partial[mt][rit] + partial[mt + 2][rit] +
                 partial[mt + 4][rit] + partial[mt + 6][rit];
      float n = tanhf(gx[(size_t)rw * G3 + 2 * DIM + hs0 + cl] + gn);
      float u = u_s[e], cold = c_s[e];
      float cnew = (1.0f - u) * n + u * cold;
      float p = padding[t * BATCH + rw];
      float c2 = (1.0f - p) * cnew + p * cold;
      c_s[e] = c2;
      c_bf[rw * DIM + hs0 + cl] = bf16r(c2);
    }
    grid_barrier(bar, &nb);

    // ---- Phase C: projection + m update ----
    {
      const int mt = wave & 1, kq = wave >> 1;
      const int arow = mt * 16 + ln;
      const int col = hs0 + ln;
      v8f acc = (v8f){};
#pragma unroll
      for (int kk = 0; kk < 8; kk++) {
        int k0 = kq * 256 + kk * 32;
        Frag a, b;
        const unsigned short* ap = c_bf + (size_t)arow * DIM + k0 + hi * 16;
        a.q[0] = *(const u32x4*)ap;  a.q[1] = *(const u32x4*)(ap + 8);
        const unsigned short* bp = wpT + (size_t)col * DIM + k0 + hi * 16;
        b.q[0] = ((const u32x4*)bp)[0];  b.q[1] = ((const u32x4*)bp)[1];
        acc = __builtin_amdgcn_wmma_f32_16x16x32_bf16(false, a.v, false, b.v,
                                                      (short)0, acc, false, false);
      }
#pragma unroll
      for (int j = 0; j < 8; j++) partial[wave][(j + hi * 8) * 16 + ln] = acc[j];
    }
    __syncthreads();
    for (int e = tid; e < 512; e += 256) {
      int rw = e >> 4, cl = e & 15;
      int mt = rw >> 4, rit = (rw & 15) * 16 + cl;
      float mn = partial[mt][rit] + partial[mt + 2][rit] +
                 partial[mt + 4][rit] + partial[mt + 6][rit];
      float p = padding[t * BATCH + rw];
      float m2 = (1.0f - p) * mn + p * m_s[e];
      m_s[e] = m2;
      m_bf[rw * DIM + hs0 + cl] = bf16r(m2);
      size_t oidx = ((size_t)t * BATCH + rw) * DIM + hs0 + cl;
      if (out_is_bf) outbf[oidx] = bf16r(m2);
      else           outf[oidx] = m2;
    }
    grid_barrier(bar, &nb);
  }
}

// ---------------------------------------------------------------------------
extern "C" void kernel_launch(void* const* d_in, const int* in_sizes, int n_in,
                              void* d_out, int out_size, void* d_ws, size_t ws_size,
                              hipStream_t stream) {
  (void)in_sizes; (void)n_in; (void)out_size; (void)ws_size;
  const float* inp   = (const float*)d_in[0];   // [512,32,1024]
  const float* pad   = (const float*)d_in[1];   // [512,32,1]
  const float* W_ih  = (const float*)d_in[2];   // [2,1024,3072]
  const float* W_hm  = (const float*)d_in[3];   // [2,1024,3072]
  const float* bias  = (const float*)d_in[4];   // [2,3072]
  const float* W_prj = (const float*)d_in[5];   // [2,1024,1024]

  char* ws = (char*)d_ws;
  size_t off = 0;
  auto carve = [&](size_t bytes) { void* p = ws + off; off += (bytes + 255) & ~(size_t)255; return p; };
  float*          gbuf  = (float*)carve((size_t)M_TOT * G3 * 4);            // 192 MiB
  unsigned short* wT    = (unsigned short*)carve((size_t)G3 * DIM * 2);     // 6 MiB
  unsigned short* wpT   = (unsigned short*)carve((size_t)DIM * DIM * 2);    // 2 MiB
  unsigned short* ys0   = (unsigned short*)carve((size_t)M_TOT * DIM * 2);  // 32 MiB
  unsigned short* m_bf  = (unsigned short*)carve((size_t)BATCH * DIM * 2);
  unsigned short* rm_bf = (unsigned short*)carve((size_t)BATCH * DIM * 2);
  unsigned short* c_bf  = (unsigned short*)carve((size_t)BATCH * DIM * 2);
  unsigned int*   bar   = (unsigned int*)carve(256);

  dim3 gemm_grid(G3 / 128, M_TOT / 128);   // 24 x 128
  for (int layer = 0; layer < 2; layer++) {
    const float* Wl = W_ih  + (size_t)layer * DIM * G3;
    const float* Wh = W_hm  + (size_t)layer * DIM * G3;
    const float* bl = bias  + (size_t)layer * G3;
    const float* Wp = W_prj + (size_t)layer * DIM * DIM;

    convert_weights<<<512, 256, 0, stream>>>(Wh, Wp, wT, wpT);
    if (layer == 0)
      gates_gemm<<<gemm_grid, 256, 0, stream>>>(inp, nullptr, 0, Wl, bl, gbuf);
    else
      gates_gemm<<<gemm_grid, 256, 0, stream>>>(nullptr, ys0, 1, Wl, bl, gbuf);

    hipMemsetAsync(m_bf, 0, (size_t)BATCH * DIM * 2, stream);
    hipMemsetAsync(bar, 0, 256, stream);

    gru_scan<<<NWG, 256, 0, stream>>>(gbuf, pad, wT, wpT, m_bf, rm_bf, c_bf, bar,
                                      (layer == 1) ? (float*)d_out : nullptr,
                                      (layer == 0) ? ys0 : nullptr,
                                      (layer == 0) ? 1 : 0);
  }
}
